// GNNEncoder_30279519436917
// MI455X (gfx1250) — matrix operand order repaired and checked by reading, not compiled, not run on hardware
//
#include <hip/hip_runtime.h>
#include <hip/hip_bf16.h>

// ---------------------------------------------------------------------------
// GCN encoder for MI455X (gfx1250, wave32, WMMA, async global->LDS).
// GEMMs run as split-bf16 (hi+lo) WMMA with fp32 accumulation:
//   A*B ~= Ahi*Bhi + Alo*Bhi + Ahi*Blo   (3x v_wmma_f32_16x16x32_bf16)
// A tiles are staged global->LDS with GLOBAL_LOAD_ASYNC_TO_LDS_B128
// (ASYNCcnt / s_wait_asynccnt path), converted to bf16 fragments once per
// block, and broadcast to all 8 waves via ds_load_b128.
// ---------------------------------------------------------------------------

typedef __attribute__((ext_vector_type(16))) __bf16 v16bf;
typedef __attribute__((ext_vector_type(8)))  float  v8f;
typedef unsigned int u32;

#define GCN_HID 256
#define GCN_IN  384
#define GCN_EPS 1e-5f
#define MAX_KT  (GCN_IN / 32)          // 12 k-tiles max (layer 1)

__device__ __forceinline__ u32 bf_rne(float f) {
  u32 u = __float_as_uint(f);
  return (u + 0x7FFFu + ((u >> 16) & 1u)) >> 16;   // round-to-nearest-even bf16
}

// pack two floats into one dword of bf16 "hi" parts and one dword of "lo" parts
__device__ __forceinline__ void cvt_pair(float a, float b, u32& hi, u32& lo) {
  u32 ha = bf_rne(a); float ra = __uint_as_float(ha << 16);
  u32 hb = bf_rne(b); float rb = __uint_as_float(hb << 16);
  u32 la = bf_rne(a - ra);
  u32 lb = bf_rne(b - rb);
  hi = ha | (hb << 16);
  lo = la | (lb << 16);
}

union Frag {
  v16bf v;
  u32   u[8];
  uint4 q[2];
};

// ---------------------------------------------------------------------------
// Pack W [K, Nc] row-major fp32 into per-lane B-fragment tiles (hi & lo bf16).
// Tile (nt, kt): 32 lanes x 16 bf16, B layout in contiguous k order (A uses the same):
//   lanes 0-15: n = lane,    K = kt*32 + 0..15   (V0 = K0,K1 ... V7 = K14,K15)
//   lanes16-31: n = lane-16, K = kt*32 + 16..31
// Tile index t = nt*ktiles + kt, lane data at u32 offset t*256 + lane*8.
// ---------------------------------------------------------------------------
__global__ void gcn_pack_w(const float* __restrict__ W, int K, int Nc,
                           u32* __restrict__ wh, u32* __restrict__ wl) {
  int tid = blockIdx.x * blockDim.x + threadIdx.x;
  int ktiles = K >> 5;
  int ntiles = Nc >> 4;
  int total = ktiles * ntiles * 32;
  if (tid >= total) return;
  int lane = tid & 31;
  int t    = tid >> 5;
  int nt   = t / ktiles;
  int kt   = t - nt * ktiles;
  int n     = nt * 16 + (lane & 15);
  int kbase = kt * 32 + ((lane >> 4) << 4);
  size_t o = (size_t)t * 256 + (size_t)lane * 8;
  for (int j = 0; j < 8; ++j) {
    float a = W[(size_t)(kbase + 2 * j)     * Nc + n];
    float b = W[(size_t)(kbase + 2 * j + 1) * Nc + n];
    u32 hi, lo;
    cvt_pair(a, b, hi, lo);
    wh[o + j] = hi;
    wl[o + j] = lo;
  }
}

// ---------------------------------------------------------------------------
// WMMA GEMM: C[M, 256] = A[M, K] @ W (pre-packed hi/lo fragments).
// Block = 256 threads = 8 waves; block covers 16 rows x 256 cols.
// Wave w owns column tiles nt0 = 2w, 2w+1 -> two v8f accumulators.
// ---------------------------------------------------------------------------
__global__ __launch_bounds__(256) void gcn_gemm_wmma(
    const float* __restrict__ A, const u32* __restrict__ wh,
    const u32* __restrict__ wl, float* __restrict__ C, int M, int K) {
  const int Nc = GCN_HID;
  __shared__ alignas(16) float sRaw[16 * GCN_IN];     // raw A tile (24 KB max)
  __shared__ alignas(16) u32   sAh[MAX_KT * 256];     // A hi fragments (12 KB)
  __shared__ alignas(16) u32   sAl[MAX_KT * 256];     // A lo fragments (12 KB)

  int tid  = threadIdx.x;
  int lane = tid & 31;
  int wave = tid >> 5;
  int m0   = blockIdx.x << 4;
  int ktiles = K >> 5;

  // ---- phase 1: async raw copy of the 16 x K fp32 A tile into LDS --------
  // one b128 (float4) per lane per issue; tracked by ASYNCcnt.
  {
    int cpr    = K >> 2;        // float4 chunks per row
    int nchunk = cpr << 4;      // 16 rows
    for (int c = tid; c < nchunk; c += 256) {
      int row  = c / cpr;
      int col4 = c - row * cpr;
      int m = m0 + row; if (m >= M) m = M - 1;
      const float* gp = A + (size_t)m * K + (col4 << 2);
      // low 32 bits of a generic LDS pointer == wave-relative LDS byte offset
      u32 loff = (u32)(uintptr_t)(&sRaw[(size_t)c << 2]);
      unsigned long long ga = (unsigned long long)(uintptr_t)gp;
      asm volatile("global_load_async_to_lds_b128 %0, %1, off"
                   :: "v"(loff), "v"(ga) : "memory");
    }
    asm volatile("s_wait_asynccnt 0" ::: "memory");
  }
  __syncthreads();

  // ---- phase 2: build A hi/lo fragments once per block --------------------
  // A fragment (16x32 bf16, contiguous k order matching B): lane m = lane&15,
  //   lanes 0-15:  K = k0+0..15
  //   lanes16-31:  K = k0+16..31
  {
    int nfrag = ktiles << 5;
    for (int f = tid; f < nfrag; f += 256) {
      int kt = f >> 5, l = f & 31;
      int abase = (l >> 4) << 4;
      const float* p = &sRaw[(size_t)(l & 15) * K + (kt << 5) + abase];
      float4 a0 = *(const float4*)(p);
      float4 a1 = *(const float4*)(p + 4);
      float4 a2 = *(const float4*)(p + 8);
      float4 a3 = *(const float4*)(p + 12);
      u32* ph = &sAh[(size_t)f << 3];
      u32* pl = &sAl[(size_t)f << 3];
      cvt_pair(a0.x, a0.y, ph[0], pl[0]);
      cvt_pair(a0.z, a0.w, ph[1], pl[1]);
      cvt_pair(a1.x, a1.y, ph[2], pl[2]);
      cvt_pair(a1.z, a1.w, ph[3], pl[3]);
      cvt_pair(a2.x, a2.y, ph[4], pl[4]);
      cvt_pair(a2.z, a2.w, ph[5], pl[5]);
      cvt_pair(a3.x, a3.y, ph[6], pl[6]);
      cvt_pair(a3.z, a3.w, ph[7], pl[7]);
    }
  }
  __syncthreads();

  // ---- phase 3: WMMA main loop -------------------------------------------
  int nt0 = wave * 2;
  v8f acc0 = {};
  v8f acc1 = {};

  for (int kt = 0; kt < ktiles; ++kt) {
    const u32* pah = &sAh[(kt << 8) + (lane << 3)];
    const u32* pal = &sAl[(kt << 8) + (lane << 3)];
    Frag ah, al;
    ah.q[0] = *(const uint4*)(pah); ah.q[1] = *(const uint4*)(pah + 4);
    al.q[0] = *(const uint4*)(pal); al.q[1] = *(const uint4*)(pal + 4);

    size_t t0 = ((size_t)nt0 * ktiles + kt) * 256 + (size_t)lane * 8;
    size_t t1 = t0 + (size_t)ktiles * 256;
    Frag bh0, bl0, bh1, bl1;
    bh0.q[0] = *(const uint4*)(wh + t0); bh0.q[1] = *(const uint4*)(wh + t0 + 4);
    bl0.q[0] = *(const uint4*)(wl + t0); bl0.q[1] = *(const uint4*)(wl + t0 + 4);
    bh1.q[0] = *(const uint4*)(wh + t1); bh1.q[1] = *(const uint4*)(wh + t1 + 4);
    bl1.q[0] = *(const uint4*)(wl + t1); bl1.q[1] = *(const uint4*)(wl + t1 + 4);
    if (kt + 1 < ktiles) {
      __builtin_prefetch((const void*)(wh + t0 + 256), 0, 1);
      __builtin_prefetch((const void*)(wl + t0 + 256), 0, 1);
    }

    acc0 = __builtin_amdgcn_wmma_f32_16x16x32_bf16(false, ah.v, false, bh0.v, (short)0, acc0, false, false);
    acc0 = __builtin_amdgcn_wmma_f32_16x16x32_bf16(false, al.v, false, bh0.v, (short)0, acc0, false, false);
    acc0 = __builtin_amdgcn_wmma_f32_16x16x32_bf16(false, ah.v, false, bl0.v, (short)0, acc0, false, false);
    acc1 = __builtin_amdgcn_wmma_f32_16x16x32_bf16(false, ah.v, false, bh1.v, (short)0, acc1, false, false);
    acc1 = __builtin_amdgcn_wmma_f32_16x16x32_bf16(false, al.v, false, bh1.v, (short)0, acc1, false, false);
    acc1 = __builtin_amdgcn_wmma_f32_16x16x32_bf16(false, ah.v, false, bl1.v, (short)0, acc1, false, false);
  }

  // C/D layout (ISA 7.12.2): VGPR r -> M = r (lanes 0-15) / r+8 (lanes 16-31)
  int col0  = nt0 * 16 + (lane & 15);
  int rbase = m0 + ((lane >> 4) << 3);
  for (int r = 0; r < 8; ++r) {
    int row = rbase + r;
    if (row < M) {
      C[(size_t)row * Nc + col0]      = acc0[r];
      C[(size_t)row * Nc + col0 + 16] = acc1[r];
    }
  }
}

// ---------------------------------------------------------------------------
// Graph aggregation + BN helper kernels
// ---------------------------------------------------------------------------
__global__ void gcn_zero(float* __restrict__ p, int n) {
  int i = blockIdx.x * blockDim.x + threadIdx.x;
  if (i < n) p[i] = 0.0f;
}

__global__ void gcn_degree(const int* __restrict__ dst, float* __restrict__ deg, int E) {
  int e = blockIdx.x * blockDim.x + threadIdx.x;
  if (e < E) atomicAdd(&deg[dst[e]], 1.0f);
}

__global__ void gcn_dinv(const float* __restrict__ deg, float* __restrict__ dinv, int n) {
  int i = blockIdx.x * blockDim.x + threadIdx.x;
  if (i < n) dinv[i] = rsqrtf(deg[i] + 1.0f);   // deg includes self-loop
}

// out[i,c] = H[i,c] * (1/deg_i) + bias[c]   (self-loop term; 1/deg = dinv^2)
__global__ void gcn_agg_init(const float* __restrict__ H, const float* __restrict__ dinv,
                             const float* __restrict__ bias, float* __restrict__ out, int n) {
  int idx = blockIdx.x * blockDim.x + threadIdx.x;
  if (idx < n) {
    int i = idx >> 8, c = idx & 255;
    float di = dinv[i];
    out[idx] = H[idx] * di * di + bias[c];
  }
}

// one wave per edge, 8 channels per lane; b128 gather + fp32 atomic scatter
__global__ __launch_bounds__(256) void gcn_agg_scatter(
    const float* __restrict__ H, const int* __restrict__ src,
    const int* __restrict__ dst, const float* __restrict__ dinv,
    float* __restrict__ out, int E) {
  int gid  = blockIdx.x * blockDim.x + threadIdx.x;
  int e    = gid >> 5;
  int lane = gid & 31;
  if (e >= E) return;
  int s = src[e], d = dst[e];
  float coef = dinv[s] * dinv[d];
  const float* hp = H   + (size_t)s * 256 + lane * 8;
  float*       op = out + (size_t)d * 256 + lane * 8;
  float4 v0 = *(const float4*)(hp);
  float4 v1 = *(const float4*)(hp + 4);
  atomicAdd(op + 0, v0.x * coef);
  atomicAdd(op + 1, v0.y * coef);
  atomicAdd(op + 2, v0.z * coef);
  atomicAdd(op + 3, v0.w * coef);
  atomicAdd(op + 4, v1.x * coef);
  atomicAdd(op + 5, v1.y * coef);
  atomicAdd(op + 6, v1.z * coef);
  atomicAdd(op + 7, v1.w * coef);
}

// per-column partial sums / sumsq over a 256-row stripe -> atomic into stats[512]
__global__ __launch_bounds__(256) void gcn_bn_stats(const float* __restrict__ H,
                                                    float* __restrict__ stats, int N) {
  int c  = threadIdx.x;
  int r0 = blockIdx.x * 256;
  int r1 = r0 + 256; if (r1 > N) r1 = N;
  float s = 0.0f, s2 = 0.0f;
  for (int r = r0; r < r1; ++r) {
    float v = H[(size_t)r * 256 + c];
    s += v; s2 += v * v;
  }
  atomicAdd(&stats[c], s);
  atomicAdd(&stats[256 + c], s2);
}

__global__ void gcn_bn_relu(float* __restrict__ H, const float* __restrict__ stats,
                            const float* __restrict__ gamma, const float* __restrict__ beta,
                            int n, float invN) {
  int idx = blockIdx.x * blockDim.x + threadIdx.x;
  if (idx < n) {
    int c = idx & 255;
    float mu  = stats[c] * invN;
    float var = stats[256 + c] * invN - mu * mu;
    float scl = gamma[c] * rsqrtf(var + GCN_EPS);
    float v = scl * (H[idx] - mu) + beta[c];
    H[idx] = fmaxf(v, 0.0f);
  }
}

// ---------------------------------------------------------------------------
extern "C" void kernel_launch(void* const* d_in, const int* in_sizes, int n_in,
                              void* d_out, int out_size, void* d_ws, size_t ws_size,
                              hipStream_t stream) {
  (void)n_in; (void)out_size; (void)ws_size;

  const float* x   = (const float*)d_in[0];
  const int*   ei  = (const int*)d_in[1];
  const float* W1  = (const float*)d_in[2];
  const float* b1  = (const float*)d_in[3];
  const float* g1  = (const float*)d_in[4];
  const float* be1 = (const float*)d_in[5];
  const float* W2  = (const float*)d_in[6];
  const float* b2  = (const float*)d_in[7];
  const float* g2  = (const float*)d_in[8];
  const float* be2 = (const float*)d_in[9];
  float* out = (float*)d_out;

  const int N = in_sizes[0] / GCN_IN;
  const int E = in_sizes[1] / 2;
  const int* src = ei;
  const int* dst = ei + E;

  // workspace carve-up (bytes): deg | dinv | stats | W1 hi/lo | W2 hi/lo | B0
  char* ws = (char*)d_ws;
  float* deg  = (float*)ws; ws += (size_t)N * 4;
  float* dinv = (float*)ws; ws += (size_t)N * 4;
  float* stats= (float*)ws; ws += 512 * 4;
  const size_t w1bytes = (size_t)(GCN_IN / 32) * (GCN_HID / 16) * 1024;   // 192 KiB
  const size_t w2bytes = (size_t)(GCN_HID / 32) * (GCN_HID / 16) * 1024;  // 128 KiB
  u32* w1h = (u32*)ws; ws += w1bytes;
  u32* w1l = (u32*)ws; ws += w1bytes;
  u32* w2h = (u32*)ws; ws += w2bytes;
  u32* w2l = (u32*)ws; ws += w2bytes;
  float* B0 = (float*)ws;                       // N * 256 fp32 (~102 MB)

  const int nElems = N * GCN_HID;
  const float invN = 1.0f / (float)N;
  const int packW1 = (GCN_IN / 32) * (GCN_HID / 16) * 32;
  const int packW2 = (GCN_HID / 32) * (GCN_HID / 16) * 32;

  // preprocessing
  gcn_zero  <<<(N + 255) / 256, 256, 0, stream>>>(deg, N);
  gcn_degree<<<(E + 255) / 256, 256, 0, stream>>>(dst, deg, E);
  gcn_dinv  <<<(N + 255) / 256, 256, 0, stream>>>(deg, dinv, N);
  gcn_pack_w<<<(packW1 + 255) / 256, 256, 0, stream>>>(W1, GCN_IN,  GCN_HID, w1h, w1l);
  gcn_pack_w<<<(packW2 + 255) / 256, 256, 0, stream>>>(W2, GCN_HID, GCN_HID, w2h, w2l);

  // ---- layer 1 ----
  gcn_gemm_wmma <<<(N + 15) / 16, 256, 0, stream>>>(x, w1h, w1l, B0, N, GCN_IN);
  gcn_agg_init  <<<(nElems + 255) / 256, 256, 0, stream>>>(B0, dinv, b1, out, nElems);
  gcn_agg_scatter<<<(E * 32 + 255) / 256, 256, 0, stream>>>(B0, src, dst, dinv, out, E);
  gcn_zero      <<<2, 256, 0, stream>>>(stats, 512);
  gcn_bn_stats  <<<(N + 255) / 256, 256, 0, stream>>>(out, stats, N);
  gcn_bn_relu   <<<(nElems + 255) / 256, 256, 0, stream>>>(out, stats, g1, be1, nElems, invN);

  // ---- layer 2 (reads `out`, ping-pongs through B0, final result in `out`) ----
  gcn_gemm_wmma <<<(N + 15) / 16, 256, 0, stream>>>(out, w2h, w2l, B0, N, GCN_HID);
  gcn_agg_init  <<<(nElems + 255) / 256, 256, 0, stream>>>(B0, dinv, b2, out, nElems);
  gcn_agg_scatter<<<(E * 32 + 255) / 256, 256, 0, stream>>>(B0, src, dst, dinv, out, E);
  gcn_zero      <<<2, 256, 0, stream>>>(stats, 512);
  gcn_bn_stats  <<<(N + 255) / 256, 256, 0, stream>>>(out, stats, N);
  gcn_bn_relu   <<<(nElems + 255) / 256, 256, 0, stream>>>(out, stats, g2, be2, nElems, invN);
}
